// ScaledDotProductAttention_22299470200953
// MI455X (gfx1250) — matrix operand order repaired and checked by reading, not compiled, not run on hardware
//
#include <hip/hip_runtime.h>

typedef __attribute__((ext_vector_type(16))) _Float16 v16h;
typedef __attribute__((ext_vector_type(8)))  _Float16 v8h;
typedef __attribute__((ext_vector_type(2)))  _Float16 v2h;
typedef __attribute__((ext_vector_type(8)))  float    v8f;

#define WMMA_F32_F16(a, b, c) \
  __builtin_amdgcn_wmma_f32_16x16x32_f16(false, (a), false, (b), (short)0, (c), false, false)

namespace {
constexpr int kB = 2, kH = 16, kS = 2048, kD = 64;
constexpr int QTILE = 128;  // query rows per workgroup (8 waves x 16 rows)
constexpr int KTILE = 32;   // key rows per inner step (== WMMA K for f16)
constexpr int KSTR  = 72;   // LDS row stride (halves) for K tile   -> conflict-free reads
constexpr int VSTR  = 40;   // LDS row stride (halves) for V^T tile -> conflict-free reads
constexpr int PSTR  = 40;   // LDS row stride (halves) for P tile
constexpr float kScaleLog2e = 0.18033688011112042f;  // (1/sqrt(64)) * log2(e)
constexpr float kNegBig = -1.0e30f;
}  // namespace

union V16 {
  v16h v;
  v8h  h[2];
};

// 16-lane butterfly reductions in pure VALU via DPP16 (no LDS bpermute).
__device__ __forceinline__ float rowmax16(float x) {
  int t;
  t = __builtin_amdgcn_update_dpp(0, __float_as_int(x), 0xB1, 0xF, 0xF, true);
  x = fmaxf(x, __int_as_float(t));
  t = __builtin_amdgcn_update_dpp(0, __float_as_int(x), 0x4E, 0xF, 0xF, true);
  x = fmaxf(x, __int_as_float(t));
  t = __builtin_amdgcn_update_dpp(0, __float_as_int(x), 0x141, 0xF, 0xF, true);
  x = fmaxf(x, __int_as_float(t));
  t = __builtin_amdgcn_update_dpp(0, __float_as_int(x), 0x140, 0xF, 0xF, true);
  x = fmaxf(x, __int_as_float(t));
  return x;
}

__device__ __forceinline__ float rowsum16(float x) {
  int t;
  t = __builtin_amdgcn_update_dpp(0, __float_as_int(x), 0xB1, 0xF, 0xF, true);
  x += __int_as_float(t);
  t = __builtin_amdgcn_update_dpp(0, __float_as_int(x), 0x4E, 0xF, 0xF, true);
  x += __int_as_float(t);
  t = __builtin_amdgcn_update_dpp(0, __float_as_int(x), 0x141, 0xF, 0xF, true);
  x += __int_as_float(t);
  t = __builtin_amdgcn_update_dpp(0, __float_as_int(x), 0x140, 0xF, 0xF, true);
  x += __int_as_float(t);
  return x;
}

__global__ __launch_bounds__(256, 2)
void fa_fwd_causal(const float* __restrict__ Qg, const float* __restrict__ Kg,
                   const float* __restrict__ Vg, float* __restrict__ Og) {
  // double-buffered K/V tiles (ping-pong) + per-wave P scratch
  __shared__ _Float16 Kl[2][KTILE * KSTR];  // K tile, row-major [32][64] f16 (padded)
  __shared__ _Float16 Vt[2][kD * VSTR];     // V tile, transposed [64][32] f16 (padded)
  __shared__ _Float16 Pl[8 * 16 * PSTR];    // per-wave prob scratch [16][32] f16

  const int tid  = threadIdx.x;
  const int wave = tid >> 5;
  const int lane = tid & 31;
  const int lh   = lane & 15;  // lane % 16
  const int hi   = lane >> 4;  // 0 = lanes 0-15, 1 = lanes 16-31

  const int qBase = blockIdx.x * QTILE;
  const int bh    = blockIdx.y;
  const size_t planeOff = (size_t)bh * kS * kD;
  const float* Qp = Qg + planeOff;
  float*       Op = Og + planeOff;

  const int qW   = qBase + wave * 16;  // first query row of this wave
  const int qRow = qW + lh;            // A-layout: lane holds row (lane % 16)

  // ---- load Q once in WMMA A layout, pre-scaled by (1/sqrt(D))*log2(e) ----
  V16 aLo, aHi;
  {
    const float* qr = Qp + (size_t)qRow * kD;
    const int base = hi * 16;  // lanes 16-31 hold K = 16..31 (contiguous, matches B)
#pragma unroll
    for (int e = 0; e < 16; ++e) {
      const int d = base + e;
      aLo.v[e] = (_Float16)(qr[d] * kScaleLog2e);
      aHi.v[e] = (_Float16)(qr[d + 32] * kScaleLog2e);
    }
  }

  v8f acc0 = {}, acc1 = {}, acc2 = {}, acc3 = {};
  float mrow[8], lrow[8];  // lrow = per-lane PARTIAL row sums (2 cols/lane)
#pragma unroll
  for (int r = 0; r < 8; ++r) { mrow[r] = kNegBig; lrow[r] = 0.0f; }

  // staging coordinates (K tile: 8 contiguous f32/thread; V: 4 f32 from 2 rows)
  const int sRow  = (tid * 8) >> 6;
  const int sCol  = (tid * 8) & 63;
  const int vPair = tid >> 4;        // key-row pair 0..15
  const int vCol  = (tid & 15) * 4;  // d offset 0..60

  // incrementing global pointers (avoid per-iter 64-bit address rebuild)
  const float* kPtr  = Kg + planeOff + (size_t)sRow * kD + sCol;
  const float* vPtrA = Vg + planeOff + (size_t)(2 * vPair) * kD + vCol;
  const float* vPtrB = Vg + planeOff + (size_t)(2 * vPair + 1) * kD + vCol;

  const int nBlocks = (qBase + QTILE) / KTILE;  // causal: keys < qBase+128

  // ---- prologue: stage block 0 into buffer 0 ----
  {
    const float4 k0 = *reinterpret_cast<const float4*>(kPtr);
    const float4 k1 = *reinterpret_cast<const float4*>(kPtr + 4);
    const float4 va = *reinterpret_cast<const float4*>(vPtrA);
    const float4 vb = *reinterpret_cast<const float4*>(vPtrB);
    kPtr += KTILE * kD; vPtrA += KTILE * kD; vPtrB += KTILE * kD;

    _Float16* kd = &Kl[0][sRow * KSTR + sCol];
    kd[0] = (_Float16)k0.x; kd[1] = (_Float16)k0.y;
    kd[2] = (_Float16)k0.z; kd[3] = (_Float16)k0.w;
    kd[4] = (_Float16)k1.x; kd[5] = (_Float16)k1.y;
    kd[6] = (_Float16)k1.z; kd[7] = (_Float16)k1.w;
    v2h p;
    p[0] = (_Float16)va.x; p[1] = (_Float16)vb.x;
    *reinterpret_cast<v2h*>(&Vt[0][(vCol + 0) * VSTR + 2 * vPair]) = p;
    p[0] = (_Float16)va.y; p[1] = (_Float16)vb.y;
    *reinterpret_cast<v2h*>(&Vt[0][(vCol + 1) * VSTR + 2 * vPair]) = p;
    p[0] = (_Float16)va.z; p[1] = (_Float16)vb.z;
    *reinterpret_cast<v2h*>(&Vt[0][(vCol + 2) * VSTR + 2 * vPair]) = p;
    p[0] = (_Float16)va.w; p[1] = (_Float16)vb.w;
    *reinterpret_cast<v2h*>(&Vt[0][(vCol + 3) * VSTR + 2 * vPair]) = p;
  }
  __syncthreads();

  for (int kb = 0; kb < nBlocks; ++kb) {
    const int kBase = kb * KTILE;
    const int buf   = kb & 1;
    const bool hasNext = (kb + 1) < nBlocks;  // block-uniform

    // ---- issue next tile's global loads NOW; consume at iteration tail ----
    float4 nk0, nk1, nva, nvb;
    if (hasNext) {
      nk0 = *reinterpret_cast<const float4*>(kPtr);
      nk1 = *reinterpret_cast<const float4*>(kPtr + 4);
      nva = *reinterpret_cast<const float4*>(vPtrA);
      nvb = *reinterpret_cast<const float4*>(vPtrB);
      kPtr += KTILE * kD; vPtrA += KTILE * kD; vPtrB += KTILE * kD;
    }

    if (kBase <= qW + 15) {  // wave-uniform: EXEC stays all-ones around WMMA
      const _Float16* Kb = Kl[buf];
      const _Float16* Vb = Vt[buf];

      // ---- scores: S(16x32) = Q(16x64) . K^T, two 16x16 tiles, K-dim split ----
      v8f s0 = {}, s1 = {};
      {
        const _Float16* kr0 = &Kb[lh * KSTR + hi * 16];         // keys kBase+0..15
        const _Float16* kr1 = &Kb[(16 + lh) * KSTR + hi * 16];  // keys kBase+16..31
        V16 b;
        b.h[0] = *(const v8h*)(kr0);      b.h[1] = *(const v8h*)(kr0 + 8);
        s0 = WMMA_F32_F16(aLo.v, b.v, s0);
        b.h[0] = *(const v8h*)(kr0 + 32); b.h[1] = *(const v8h*)(kr0 + 40);
        s0 = WMMA_F32_F16(aHi.v, b.v, s0);
        b.h[0] = *(const v8h*)(kr1);      b.h[1] = *(const v8h*)(kr1 + 8);
        s1 = WMMA_F32_F16(aLo.v, b.v, s1);
        b.h[0] = *(const v8h*)(kr1 + 32); b.h[1] = *(const v8h*)(kr1 + 40);
        s1 = WMMA_F32_F16(aHi.v, b.v, s1);
      }

      // ---- causal mask (only on diagonal-crossing blocks) ----
      float x0[8], x1[8];
      const bool needMask = (kBase + 31) > qW;  // wave-uniform
      if (needMask) {
#pragma unroll
        for (int r = 0; r < 8; ++r) {
          const int q = qW + r + hi * 8;  // C layout: VGPR r <-> row r + hi*8
          const int n = kBase + lh;
          x0[r] = (n      <= q) ? s0[r] : kNegBig;
          x1[r] = (n + 16 <= q) ? s1[r] : kNegBig;
        }
      } else {
#pragma unroll
        for (int r = 0; r < 8; ++r) { x0[r] = s0[r]; x1[r] = s1[r]; }
      }

      // ---- online softmax: DPP row-max, raw v_exp_f32, partial row sums ----
#pragma unroll
      for (int r = 0; r < 8; ++r) {
        const float mx   = rowmax16(fmaxf(x0[r], x1[r]));
        const float mnew = fmaxf(mrow[r], mx);
        const float corr = __builtin_amdgcn_exp2f(mrow[r] - mnew);
        mrow[r] = mnew;
        const float p0 = __builtin_amdgcn_exp2f(x0[r] - mnew);
        const float p1 = __builtin_amdgcn_exp2f(x1[r] - mnew);
        lrow[r] = lrow[r] * corr + (p0 + p1);
        acc0[r] *= corr; acc1[r] *= corr;
        acc2[r] *= corr; acc3[r] *= corr;
        _Float16* pr = &Pl[(wave * 16 + r + hi * 8) * PSTR];
        pr[lh]      = (_Float16)p0;
        pr[16 + lh] = (_Float16)p1;
      }
      // intra-wave LDS store->load ordering (CDNA5 split DS counter)
      asm volatile("s_wait_dscnt 0" ::: "memory");

      // ---- reload P in A layout, then O += P(16x32) . V(32x64) ----
      V16 pa;
      {
        const _Float16* pw = &Pl[(wave * 16 + lh) * PSTR + hi * 16];
        pa.h[0] = *(const v8h*)(pw);
        pa.h[1] = *(const v8h*)(pw + 8);
      }
      {
        V16 b;
        const _Float16* v0r = &Vb[(0 * 16 + lh) * VSTR + hi * 16];
        b.h[0] = *(const v8h*)(v0r); b.h[1] = *(const v8h*)(v0r + 8);
        acc0 = WMMA_F32_F16(pa.v, b.v, acc0);
        const _Float16* v1r = &Vb[(1 * 16 + lh) * VSTR + hi * 16];
        b.h[0] = *(const v8h*)(v1r); b.h[1] = *(const v8h*)(v1r + 8);
        acc1 = WMMA_F32_F16(pa.v, b.v, acc1);
        const _Float16* v2r = &Vb[(2 * 16 + lh) * VSTR + hi * 16];
        b.h[0] = *(const v8h*)(v2r); b.h[1] = *(const v8h*)(v2r + 8);
        acc2 = WMMA_F32_F16(pa.v, b.v, acc2);
        const _Float16* v3r = &Vb[(3 * 16 + lh) * VSTR + hi * 16];
        b.h[0] = *(const v8h*)(v3r); b.h[1] = *(const v8h*)(v3r + 8);
        acc3 = WMMA_F32_F16(pa.v, b.v, acc3);
      }
    }

    // ---- tail: convert + store next tile into the other buffer ----
    if (hasNext) {
      const int nbuf = buf ^ 1;
      _Float16* kd = &Kl[nbuf][sRow * KSTR + sCol];
      kd[0] = (_Float16)nk0.x; kd[1] = (_Float16)nk0.y;
      kd[2] = (_Float16)nk0.z; kd[3] = (_Float16)nk0.w;
      kd[4] = (_Float16)nk1.x; kd[5] = (_Float16)nk1.y;
      kd[6] = (_Float16)nk1.z; kd[7] = (_Float16)nk1.w;
      v2h p;
      p[0] = (_Float16)nva.x; p[1] = (_Float16)nvb.x;
      *reinterpret_cast<v2h*>(&Vt[nbuf][(vCol + 0) * VSTR + 2 * vPair]) = p;
      p[0] = (_Float16)nva.y; p[1] = (_Float16)nvb.y;
      *reinterpret_cast<v2h*>(&Vt[nbuf][(vCol + 1) * VSTR + 2 * vPair]) = p;
      p[0] = (_Float16)nva.z; p[1] = (_Float16)nvb.z;
      *reinterpret_cast<v2h*>(&Vt[nbuf][(vCol + 2) * VSTR + 2 * vPair]) = p;
      p[0] = (_Float16)nva.w; p[1] = (_Float16)nvb.w;
      *reinterpret_cast<v2h*>(&Vt[nbuf][(vCol + 3) * VSTR + 2 * vPair]) = p;
    }
    __syncthreads();
  }

  // ---- epilogue: finish row sums (deferred reduction), normalize, store ----
#pragma unroll
  for (int r = 0; r < 8; ++r) {
    const float inv = __builtin_amdgcn_rcpf(rowsum16(lrow[r]));
    const int q = qW + r + hi * 8;
    float* orow = Op + (size_t)q * kD + lh;
    orow[0]  = acc0[r] * inv;
    orow[16] = acc1[r] * inv;
    orow[32] = acc2[r] * inv;
    orow[48] = acc3[r] * inv;
  }
}

extern "C" void kernel_launch(void* const* d_in, const int* in_sizes, int n_in,
                              void* d_out, int out_size, void* d_ws, size_t ws_size,
                              hipStream_t stream) {
  (void)in_sizes; (void)n_in; (void)out_size; (void)d_ws; (void)ws_size;
  const float* Q = (const float*)d_in[0];
  const float* K = (const float*)d_in[1];
  const float* V = (const float*)d_in[2];
  float* O = (float*)d_out;
  dim3 grid(kS / QTILE, kB * kH);
  dim3 block(256);
  hipLaunchKernelGGL(fa_fwd_causal, grid, block, 0, stream, Q, K, V, O);
}
